// EquiDeformableAttn_3410204033228
// MI455X (gfx1250) — matrix operand order repaired and checked by reading, not compiled, not run on hardware
//
#include <hip/hip_runtime.h>
#include <hip/hip_bf16.h>

typedef __attribute__((ext_vector_type(16))) __bf16 v16bf;
typedef __attribute__((ext_vector_type(8)))  __bf16 v8bf;
typedef __attribute__((ext_vector_type(8)))  float  v8f;

#define B_  8
#define N_  4096
#define C_  128
#define S_  8
#define H_  64
#define HW_ (H_ * H_)

// Packed B-operand buffer: [nt(8)][kb(4)][lane(32)][j(16)] bf16, 32B-aligned per lane.
#define PACKED_ELEMS (8 * 4 * 32 * 16)

// ---------------- WMMA fragment builders (CDNA5 16x16x32 bf16 layouts) ----------------
// A (16x32, MxK): lanes 0-15 -> M=lane, halves[0..7]=K0..7, halves[8..15]=K16..23
//                 lanes 16-31 -> M=lane-16, halves = K8..15 / K24..31
// Two contiguous 16B LDS loads per fragment.
static __device__ inline v16bf load_a_frag(const __bf16* tile, int kb, int lane) {
  const int row = lane & 15;
  const int khi = (lane >> 4) << 3;            // 0 or 8
  const __bf16* r = tile + row * C_ + kb * 32 + khi;
  v8bf lo = *(const v8bf*)(r);                 // K = kb*32 + khi + 0..7
  v8bf hi = *(const v8bf*)(r + 16);            // K = kb*32 + 16 + khi + 0..7
  return __builtin_shufflevector(lo, hi, 0, 1, 2, 3, 4, 5, 6, 7,
                                         8, 9, 10, 11, 12, 13, 14, 15);
}

// B fragment: one 32B contiguous load from the pre-packed weight buffer.
static __device__ inline v16bf load_b_packed(const __bf16* Wp, int nt, int kb, int lane) {
  return *(const v16bf*)(Wp + ((((nt << 2) + kb) * 32 + lane) << 4));
}

// ---------------- bilinear setup (grid_sample, border, align_corners=True) ----------------
static __device__ inline void bil_setup(float cx, float cy,
                                        int& o00, int& o01, int& o10, int& o11,
                                        float& w00, float& w01, float& w10, float& w11) {
  float fx = fminf(fmaxf((cx + 1.f) * 0.5f * (float)(H_ - 1), 0.f), (float)(H_ - 1));
  float fy = fminf(fmaxf((cy + 1.f) * 0.5f * (float)(H_ - 1), 0.f), (float)(H_ - 1));
  int x0 = (int)fx, y0 = (int)fy;
  int x1 = x0 + 1; if (x1 > H_ - 1) x1 = H_ - 1;
  int y1 = y0 + 1; if (y1 > H_ - 1) y1 = H_ - 1;
  float wx = fx - (float)x0, wy = fy - (float)y0;
  w00 = (1.f - wx) * (1.f - wy); w01 = wx * (1.f - wy);
  w10 = (1.f - wx) * wy;         w11 = wx * wy;
  o00 = y0 * H_ + x0; o01 = y0 * H_ + x1;
  o10 = y1 * H_ + x0; o11 = y1 * H_ + x1;
}

// ---------------- prep: fused weight matrices, written fragment-packed ----------------
// Wqk[K][N]    = sqrt(C) * sum_j Wq[K][j] * Wk[N][j]         (k never materialized)
// Wfused[K][N] = sum_j Wv[K][j] * Wout[j][N]                 (v never materialized)
// qbias[N]     = sqrt(C) * sum_j bq[j] * Wk[N][j]
// bfused[N]    = sum_j bv[j] * Wout[j][N] + bout[N]
// Packed index for element (K, N):
//   kb = K>>5, krem = K&31, nt = N>>4, nl = N&15
//   lane = ((krem>>3)&1) ? 16+nl : nl ;  j = (krem & 7) | ((krem>>4)<<3)
__global__ void __launch_bounds__(C_)
prep_kernel(const float* __restrict__ Wq, const float* __restrict__ bq,
            const float* __restrict__ Wk, const float* __restrict__ Wv,
            const float* __restrict__ Wout, const float* __restrict__ bout,
            const float* __restrict__ bv,
            __bf16* __restrict__ WqkP, __bf16* __restrict__ WfusP,
            float* __restrict__ qbias, float* __restrict__ bfus) {
  const int K = blockIdx.x, Ncol = threadIdx.x;
  const float sqrtC = 11.313708498984761f;
  float a0 = 0.f, a1 = 0.f;
  for (int j = 0; j < C_; ++j) {
    a0 += Wq[K * C_ + j] * Wk[Ncol * C_ + j];
    a1 += Wv[K * C_ + j] * Wout[j * C_ + Ncol];
  }
  const int kb = K >> 5, krem = K & 31;
  const int nt = Ncol >> 4, nl = Ncol & 15;
  const int lane = ((krem >> 3) & 1) ? (16 + nl) : nl;
  const int j16  = (krem & 7) | ((krem >> 4) << 3);
  const int idx  = ((((nt << 2) + kb) * 32 + lane) << 4) + j16;
  WqkP[idx]  = (__bf16)(a0 * sqrtC);
  WfusP[idx] = (__bf16)a1;
  if (K == 0) {
    float qb = 0.f, bf = 0.f;
    for (int j = 0; j < C_; ++j) {
      qb += bq[j] * Wk[Ncol * C_ + j];
      bf += bv[j] * Wout[j * C_ + Ncol];
    }
    qbias[Ncol] = qb * sqrtC;
    bfus[Ncol]  = bf + bout[Ncol];
  }
}

// ---------------- kernel1: sample feature, offsets -> aux points, qt = feat@Wqk + qbias ----
__global__ void __launch_bounds__(128)
kernel1(const float* __restrict__ qp,
        const float* __restrict__ c_xz, const float* __restrict__ c_xy,
        const float* __restrict__ c_yz,
        const float* __restrict__ Woff, const float* __restrict__ boff,
        const __bf16* __restrict__ WqkP, const float* __restrict__ qbias,
        float* __restrict__ feat_g, float* __restrict__ qt_g,
        float* __restrict__ aux_g) {
  __shared__ float  featf[16][C_];
  __shared__ alignas(32) __bf16 featb[16][C_];
  __shared__ int    sidx[16][3][4];
  __shared__ float  swt[16][3][4];

  const int t    = threadIdx.x;
  const int tile = blockIdx.x;
  const int b    = tile >> 8;                 // N_/16 = 256 tiles per batch
  const int n0   = (tile & 255) << 4;
  const long bn0 = (long)b * N_ + n0;

  if (t < 48) {                               // per (query, plane) bilinear setup
    int q = t / 3, pl = t - 3 * q;
    const float* p = qp + (bn0 + q) * 3;
    float px = p[0], py = p[1], pz = p[2];
    float cx = (pl == 2) ? py : px;
    float cy = (pl == 1) ? py : pz;
    int o0, o1, o2, o3; float w0, w1, w2, w3;
    bil_setup(cx, cy, o0, o1, o2, o3, w0, w1, w2, w3);
    sidx[q][pl][0] = o0; sidx[q][pl][1] = o1; sidx[q][pl][2] = o2; sidx[q][pl][3] = o3;
    swt[q][pl][0] = w0;  swt[q][pl][1] = w1;  swt[q][pl][2] = w2;  swt[q][pl][3] = w3;
  }
  __syncthreads();

  const int c = t;                            // one channel per thread
  const float* base0 = c_xz + (size_t)(b * C_ + c) * HW_;
  const float* base1 = c_xy + (size_t)(b * C_ + c) * HW_;
  const float* base2 = c_yz + (size_t)(b * C_ + c) * HW_;
  __builtin_prefetch(base0, 0, 0);
  __builtin_prefetch(base1, 0, 0);
  __builtin_prefetch(base2, 0, 0);
  for (int q = 0; q < 16; ++q) {
    float acc;
    acc  = swt[q][0][0] * base0[sidx[q][0][0]] + swt[q][0][1] * base0[sidx[q][0][1]]
         + swt[q][0][2] * base0[sidx[q][0][2]] + swt[q][0][3] * base0[sidx[q][0][3]];
    acc += swt[q][1][0] * base1[sidx[q][1][0]] + swt[q][1][1] * base1[sidx[q][1][1]]
         + swt[q][1][2] * base1[sidx[q][1][2]] + swt[q][1][3] * base1[sidx[q][1][3]];
    acc += swt[q][2][0] * base2[sidx[q][2][0]] + swt[q][2][1] * base2[sidx[q][2][1]]
         + swt[q][2][2] * base2[sidx[q][2][2]] + swt[q][2][3] * base2[sidx[q][2][3]];
    featf[q][c] = acc;
    featb[q][c] = (__bf16)acc;
    feat_g[(bn0 + q) * C_ + c] = acc;
  }
  __syncthreads();

  // deformable offsets: 16 queries * 24 outputs = 384 dots of length 128
  for (int r = 0; r < 3; ++r) {
    int o = t + r * 128;
    if (o < 16 * 3 * S_) {
      int q = o / (3 * S_);
      int j = o - q * (3 * S_);
      float acc = boff[j];
      for (int d = 0; d < C_; ++d) acc += featf[q][d] * Woff[d * (3 * S_) + j];
      int k = j % 3;                          // j = s*3 + k
      aux_g[(bn0 + q) * (3 * S_) + j] = acc + qp[(bn0 + q) * 3 + k];
    }
  }

  // qt = featb @ Wqk + qbias   (WMMA bf16: 8 N-tiles x 4 K-steps, 4 waves x 2 tiles)
  const int wave = t >> 5, lane = t & 31;
  for (int nti = 0; nti < 2; ++nti) {
    const int nt = wave + nti * 4;
    v8f acc = {};
#pragma unroll
    for (int kb = 0; kb < 4; ++kb) {
      v16bf a  = load_a_frag(&featb[0][0], kb, lane);
      v16bf bb = load_b_packed(WqkP, nt, kb, lane);
      acc = __builtin_amdgcn_wmma_f32_16x16x32_bf16(false, a, false, bb,
                                                    (short)0, acc, false, false);
    }
    const int col   = nt * 16 + (lane & 15);
    const int mbase = (lane >> 4) << 3;
    const float qb  = qbias[col];
#pragma unroll
    for (int r = 0; r < 8; ++r)
      qt_g[(bn0 + r + mbase) * C_ + col] = acc[r] + qb;
  }
}

// ---------------- kernel2: aux sampling + attention + fused output projection -----------
__global__ void __launch_bounds__(128)
kernel2(const float* __restrict__ aux_g,
        const float* __restrict__ c_xz, const float* __restrict__ c_xy,
        const float* __restrict__ c_yz,
        const float* __restrict__ qt_g, const float* __restrict__ feat_g,
        const __bf16* __restrict__ WfusP, const float* __restrict__ bfus,
        float* __restrict__ out_g) {
  __shared__ alignas(32) __bf16 auxb[128][C_]; // 32 KB: 16 queries x 8 samples x 128 ch
  __shared__ float  qts[16][C_];               // 8 KB
  __shared__ int    sidx[128][3][4];           // 6 KB
  __shared__ float  swt[128][3][4];            // 6 KB
  __shared__ float  attn[128];
  __shared__ alignas(32) __bf16 mixedb[16][C_];// 4 KB

  const int t    = threadIdx.x;
  const int tile = blockIdx.x;
  const int b    = tile >> 8;
  const int n0   = (tile & 255) << 4;
  const long bn0 = (long)b * N_ + n0;

  {                                           // per-point bilinear setup: point p = t
    const int p = t, q = p >> 3;
    const float* ap = aux_g + (bn0 + q) * (3 * S_) + (p & 7) * 3;
    float px = ap[0], py = ap[1], pz = ap[2];
#pragma unroll
    for (int pl = 0; pl < 3; ++pl) {
      float cx = (pl == 2) ? py : px;
      float cy = (pl == 1) ? py : pz;
      int o0, o1, o2, o3; float w0, w1, w2, w3;
      bil_setup(cx, cy, o0, o1, o2, o3, w0, w1, w2, w3);
      sidx[p][pl][0] = o0; sidx[p][pl][1] = o1; sidx[p][pl][2] = o2; sidx[p][pl][3] = o3;
      swt[p][pl][0] = w0;  swt[p][pl][1] = w1;  swt[p][pl][2] = w2;  swt[p][pl][3] = w3;
    }
  }
  for (int i = 0; i < 16; ++i) {              // stage qt tile into LDS
    int idx = t + i * 128;
    qts[idx >> 7][idx & 127] = qt_g[(bn0 + (idx >> 7)) * C_ + (idx & 127)];
  }
  __syncthreads();

  const int c = t;
  const float* base0 = c_xz + (size_t)(b * C_ + c) * HW_;
  const float* base1 = c_xy + (size_t)(b * C_ + c) * HW_;
  const float* base2 = c_yz + (size_t)(b * C_ + c) * HW_;
  __builtin_prefetch(base0, 0, 0);
  __builtin_prefetch(base1, 0, 0);
  __builtin_prefetch(base2, 0, 0);
#pragma unroll 2
  for (int p = 0; p < 128; ++p) {
    float acc;
    acc  = swt[p][0][0] * base0[sidx[p][0][0]] + swt[p][0][1] * base0[sidx[p][0][1]]
         + swt[p][0][2] * base0[sidx[p][0][2]] + swt[p][0][3] * base0[sidx[p][0][3]];
    acc += swt[p][1][0] * base1[sidx[p][1][0]] + swt[p][1][1] * base1[sidx[p][1][1]]
         + swt[p][1][2] * base1[sidx[p][1][2]] + swt[p][1][3] * base1[sidx[p][1][3]];
    acc += swt[p][2][0] * base2[sidx[p][2][0]] + swt[p][2][1] * base2[sidx[p][2][1]]
         + swt[p][2][2] * base2[sidx[p][2][2]] + swt[p][2][3] * base2[sidx[p][2][3]];
    auxb[p][c] = (__bf16)acc;
  }
  __syncthreads();

  {                                           // sim[p] = qt[n(p)] . aux[p]
    const int p = t, q = p >> 3;
    float s = 0.f;
    for (int d = 0; d < C_; ++d) s += qts[q][d] * (float)auxb[p][d];
    attn[p] = s;
  }
  __syncthreads();

  if (t < 16) {                               // softmax over S=8 per query
    float m = attn[t * 8];
#pragma unroll
    for (int s = 1; s < 8; ++s) m = fmaxf(m, attn[t * 8 + s]);
    float e[8], sum = 0.f;
#pragma unroll
    for (int s = 0; s < 8; ++s) { e[s] = __expf(attn[t * 8 + s] - m); sum += e[s]; }
    float inv = 1.f / sum;
#pragma unroll
    for (int s = 0; s < 8; ++s) attn[t * 8 + s] = e[s] * inv;
  }
  __syncthreads();

  for (int q = 0; q < 16; ++q) {              // mixed[q][c] = sum_s attn * aux
    float acc = 0.f;
#pragma unroll
    for (int s = 0; s < 8; ++s) acc += attn[q * 8 + s] * (float)auxb[q * 8 + s][c];
    mixedb[q][c] = (__bf16)acc;
  }
  __syncthreads();

  // out = mixed @ Wfused + bfused + feat   (WMMA bf16)
  const int wave = t >> 5, lane = t & 31;
  for (int nti = 0; nti < 2; ++nti) {
    const int nt = wave + nti * 4;
    v8f acc = {};
#pragma unroll
    for (int kb = 0; kb < 4; ++kb) {
      v16bf a  = load_a_frag(&mixedb[0][0], kb, lane);
      v16bf bb = load_b_packed(WfusP, nt, kb, lane);
      acc = __builtin_amdgcn_wmma_f32_16x16x32_bf16(false, a, false, bb,
                                                    (short)0, acc, false, false);
    }
    const int col   = nt * 16 + (lane & 15);
    const int mbase = (lane >> 4) << 3;
    const float bf  = bfus[col];
#pragma unroll
    for (int r = 0; r < 8; ++r) {
      const long row = bn0 + r + mbase;
      out_g[row * C_ + col] = acc[r] + bf + feat_g[row * C_ + col];
    }
  }
}

// ---------------- launch ----------------
extern "C" void kernel_launch(void* const* d_in, const int* in_sizes, int n_in,
                              void* d_out, int out_size, void* d_ws, size_t ws_size,
                              hipStream_t stream) {
  (void)in_sizes; (void)n_in; (void)out_size; (void)ws_size;
  const float* qp   = (const float*)d_in[0];
  const float* cxz  = (const float*)d_in[1];
  const float* cxy  = (const float*)d_in[2];
  const float* cyz  = (const float*)d_in[3];
  const float* Wq   = (const float*)d_in[4];
  const float* bq   = (const float*)d_in[5];
  const float* Wk   = (const float*)d_in[6];
  // d_in[7] = bk: constant over s -> softmax-invariant, provably drops out
  const float* Wv   = (const float*)d_in[8];
  const float* bv   = (const float*)d_in[9];
  const float* Wout = (const float*)d_in[10];
  const float* bout = (const float*)d_in[11];
  const float* Woff = (const float*)d_in[12];
  const float* boff = (const float*)d_in[13];

  const size_t BN = (size_t)B_ * N_;
  char* ws = (char*)d_ws;
  float*  feat_g = (float*)ws;  ws += BN * C_ * sizeof(float);
  float*  qt_g   = (float*)ws;  ws += BN * C_ * sizeof(float);
  float*  aux_g  = (float*)ws;  ws += BN * 3 * S_ * sizeof(float);
  __bf16* WqkP   = (__bf16*)ws; ws += (size_t)PACKED_ELEMS * sizeof(__bf16);
  __bf16* WfusP  = (__bf16*)ws; ws += (size_t)PACKED_ELEMS * sizeof(__bf16);
  float*  qbias  = (float*)ws;  ws += C_ * sizeof(float);
  float*  bfus   = (float*)ws;  ws += C_ * sizeof(float);

  prep_kernel<<<C_, C_, 0, stream>>>(Wq, bq, Wk, Wv, Wout, bout, bv,
                                     WqkP, WfusP, qbias, bfus);
  kernel1<<<dim3(BN / 16), dim3(128), 0, stream>>>(qp, cxz, cxy, cyz, Woff, boff,
                                                   WqkP, qbias, feat_g, qt_g, aux_g);
  kernel2<<<dim3(BN / 16), dim3(128), 0, stream>>>(aux_g, cxz, cxy, cyz, qt_g, feat_g,
                                                   WfusP, bfus, (float*)d_out);
}
